// LW_IRST_ablation_CA_Fuse_mamba_27384711479320
// MI455X (gfx1250) — matrix operand order repaired and checked by reading, not compiled, not run on hardware
//
#include <hip/hip_runtime.h>
#include <math.h>

#define Bn   4
#define Cc   96
#define Hh   48
#define Wd   48
#define HW   2304
#define BHW  9216
#define Dd   192
#define NST  16
#define DTR  6
#define Kk   4
#define Ll   2304
#define NCOLS 38   // DT_RANK + 2*N_STATE
#define NPAD  48   // padded to 3 WMMA tiles
#define NCH  16    // scan chunks
#define CLEN 144   // Ll / NCH

typedef __attribute__((ext_vector_type(16))) _Float16 v16h;
typedef __attribute__((ext_vector_type(8)))  _Float16 v8h;
typedef __attribute__((ext_vector_type(8)))  float    v8f;

__device__ __forceinline__ float rcp_(float x){ return __builtin_amdgcn_rcpf(x); }
__device__ __forceinline__ float sigmoidf_(float x){ return rcp_(1.f+__expf(-x)); }
__device__ __forceinline__ float siluf_(float x){ return x*sigmoidf_(x); }
__device__ __forceinline__ float softplusf_(float x){
  return (x > 20.f) ? x : __logf(1.f + __expf(x));
}

// ---------------------------------------------------------------------------
// 64x16 output per wave (4 M-subtiles): out[m][n] = sum_k A[m][k] * W[n][k]
// A: f16 [*, lda] row-major; W: f16 [*, ldw] row-major ([out,in] = B^T, so
// per-lane fragment loads are contiguous 16B). CDNA5 ISA 7.12.2 layouts.
// B fragment loaded once per k-step, shared by 4 independent WMMA chains.
// ---------------------------------------------------------------------------
__device__ __forceinline__ void wmma_rowcol4(const _Float16* __restrict__ A,
                                             const _Float16* __restrict__ W,
                                             int lda, int ldw, int kdim, int lane,
                                             v8f acc[4])
{
  const int mr = lane & 15;                 // row (A) / col (B) within tile
  const int ao = (lane < 16) ? 0 : 8;       // A: lanes 16-31 hold K+8 halves
  const int bo = (lane < 16) ? 0 : 8;       // B: lanes 16-31 hold K+8 halves
  for (int k0 = 0; k0 < kdim; k0 += 32) {
    const _Float16* bp = W + (size_t)mr*ldw + k0 + bo;
    v8h blo = *(const v8h*)(bp);            // K = k0+bo   .. +7
    v8h bhi = *(const v8h*)(bp + 16);       // K = k0+bo+16.. +23
    v16h bf;
    #pragma unroll
    for (int h=0; h<8; ++h){ bf[h]=blo[h]; bf[8+h]=bhi[h]; }
    #pragma unroll
    for (int t=0; t<4; ++t){
      const _Float16* ap = A + (size_t)(t*16 + mr)*lda + k0 + ao;
      v8h alo = *(const v8h*)(ap);          // K = k0+ao   .. +7
      v8h ahi = *(const v8h*)(ap + 16);     // K = k0+ao+16.. +23
      v16h af;
      #pragma unroll
      for (int h=0; h<8; ++h){ af[h]=alo[h]; af[8+h]=ahi[h]; }
      acc[t] = __builtin_amdgcn_wmma_f32_16x16x32_f16(false, af, false, bf,
                                                      (short)0, acc[t], false, false);
    }
  }
}

#define ACC4_INIT v8f z0_={0.f,0.f,0.f,0.f,0.f,0.f,0.f,0.f}; v8f acc[4]={z0_,z0_,z0_,z0_}

// ---- K0: pack weights to f16 (x_proj zero-padded 38 -> 48 rows) -----------
__global__ void k_wpack(const float* __restrict__ inp, const float* __restrict__ xp,
                        const float* __restrict__ op,
                        _Float16* inp_h, _Float16* xp_h, _Float16* op_h){
  int i = blockIdx.x*256 + threadIdx.x;
  if (i < 2*Dd*Cc) inp_h[i] = (_Float16)inp[i];
  if (i < Kk*NPAD*Dd){
    int k = i/(NPAD*Dd); int rem = i%(NPAD*Dd); int c = rem/Dd; int d = rem%Dd;
    xp_h[i] = (_Float16)((c < NCOLS) ? xp[(k*NCOLS + c)*Dd + d] : 0.f);
  }
  if (i < Cc*Dd) op_h[i] = (_Float16)op[i];
}

// ---- K1: s[b,c] = mean_HW x -----------------------------------------------
__global__ void k_mean(const float* __restrict__ x, float* s){
  int bc = blockIdx.x;
  const float* p = x + (size_t)bc*HW;
  float acc = 0.f;
  for (int i=threadIdx.x; i<HW; i+=256) acc += p[i];
  __shared__ float sm[256];
  sm[threadIdx.x]=acc; __syncthreads();
  for (int o=128; o>0; o>>=1){ if (threadIdx.x<o) sm[threadIdx.x]+=sm[threadIdx.x+o]; __syncthreads(); }
  if (!threadIdx.x) s[bc] = sm[0]*(1.f/HW);
}

// ---- K2: channel-attention MLP (96 -> 6 -> 96) ----------------------------
__global__ void k_camlp(const float* __restrict__ s, const float* __restrict__ w1,
                        const float* __restrict__ w2, float* a){
  __shared__ float ss[Bn*Cc];
  __shared__ float tt[Bn*6];
  for (int i=threadIdx.x; i<Bn*Cc; i+=blockDim.x) ss[i]=s[i];
  __syncthreads();
  if (threadIdx.x < Bn*6){
    int b=threadIdx.x/6, r=threadIdx.x%6;
    float acc=0.f;
    for (int c=0;c<Cc;c++) acc += ss[b*Cc+c]*w1[r*Cc+c];
    tt[b*6+r] = fmaxf(acc, 0.f);
  }
  __syncthreads();
  for (int i=threadIdx.x; i<Bn*Cc; i+=blockDim.x){
    int b=i/Cc, c=i%Cc;
    float acc=0.f;
    for (int r=0;r<6;r++) acc += tt[b*6+r]*w2[c*6+r];
    a[i] = sigmoidf_(acc);
  }
}

// ---- K3: x*att -> LayerNorm(C) -> f16 pack [BHW,96] (1 wave / row) --------
__global__ void k_ln_pack(const float* __restrict__ x, const float* __restrict__ a,
                          const float* __restrict__ g, const float* __restrict__ bta,
                          _Float16* xn_h){
  int row  = blockIdx.x*4 + (threadIdx.x>>5);   // b*HW + hw
  int lane = threadIdx.x & 31;
  int b = row / HW, hw = row % HW;
  float v[3];
  #pragma unroll
  for (int i=0;i<3;i++){
    int c = lane + i*32;
    v[i] = x[((size_t)(b*Cc+c))*HW + hw] * a[b*Cc+c];
  }
  float s1=v[0]+v[1]+v[2], s2=v[0]*v[0]+v[1]*v[1]+v[2]*v[2];
  for (int o=16;o>0;o>>=1){ s1+=__shfl_xor(s1,o,32); s2+=__shfl_xor(s2,o,32); }
  float m  = s1*(1.f/Cc);
  float rs = rsqrtf(s2*(1.f/Cc) - m*m + 1e-5f);
  #pragma unroll
  for (int i=0;i<3;i++){
    int c = lane + i*32;
    xn_h[(size_t)row*Cc + c] = (_Float16)((v[i]-m)*rs*g[c] + bta[c]);
  }
}

// ---- K4: in_proj GEMM [9216,96]x[384,96]^T; xc + silu(z), channel-last ----
__global__ void k_inproj(const _Float16* __restrict__ A, const _Float16* __restrict__ W,
                         float* xc_pre, float* zs){
  int lane = threadIdx.x & 31;
  int tM4 = blockIdx.x;                         // 64-row group
  int tN  = blockIdx.y*4 + (threadIdx.x>>5);
  ACC4_INIT;
  wmma_rowcol4(A + (size_t)tM4*64*Cc, W + (size_t)tN*16*Cc, Cc, Cc, Cc, lane, acc);
  int n = tN*16 + (lane&15);
  int roff = (lane<16)?0:8;
  if (tN < Dd/16){                              // wave-uniform branch: xc side
    #pragma unroll
    for (int t=0;t<4;t++){
      int mbase = tM4*64 + t*16 + roff;
      #pragma unroll
      for (int r=0;r<8;r++)
        xc_pre[(size_t)(mbase+r)*Dd + n] = acc[t][r];      // [b][hw][d]
    }
  } else {                                      // z side -> silu gate
    int nz = n - Dd;
    #pragma unroll
    for (int t=0;t<4;t++){
      int mbase = tM4*64 + t*16 + roff;
      #pragma unroll
      for (int r=0;r<8;r++)
        zs[(size_t)(mbase+r)*Dd + nz] = siluf_(acc[t][r]);
    }
  }
}

// ---- K5: depthwise 3x3 conv + bias + SiLU, channel-last -------------------
__global__ void k_dwconv(const float* __restrict__ xc, const float* __restrict__ cw,
                         const float* __restrict__ cb, float* out){
  int idx = blockIdx.x*256 + threadIdx.x;       // (b*HW + hw)*Dd + d
  if (idx >= Bn*HW*Dd) return;
  int d  = idx % Dd;
  int hw = (idx / Dd) % HW;
  int b  = idx / (Dd*HW);
  int h = hw / Wd, w = hw % Wd;
  float acc = cb[d];
  const float* wp = cw + d*9;
  const float* ip = xc + (size_t)b*HW*Dd + d;
  #pragma unroll
  for (int dy=-1; dy<=1; dy++){
    int hh=h+dy; if ((unsigned)hh>=Hh) continue;
    #pragma unroll
    for (int dx=-1; dx<=1; dx++){
      int ww=w+dx; if ((unsigned)ww>=Wd) continue;
      acc += wp[(dy+1)*3+(dx+1)] * ip[(size_t)(hh*Wd+ww)*Dd];
    }
  }
  out[idx] = siluf_(acc);
}

// ---- K6: pack 4 scan directions, f16, layout [k][b][l][d] -----------------
__global__ void k_pack_xs(const float* __restrict__ xcs, _Float16* xs_h){
  size_t idx = (size_t)blockIdx.x*256 + threadIdx.x;
  if (idx >= (size_t)Kk*Bn*Ll*Dd) return;
  int d = (int)(idx % Dd);
  int l = (int)((idx / Dd) % Ll);
  int b = (int)((idx / ((size_t)Dd*Ll)) % Bn);
  int k = (int)( idx / ((size_t)Dd*Ll*Bn));
  int l2 = (k>=2) ? (Ll-1-l) : l;
  int pos;
  if (k & 1){ int w = l2 / Hh, h = l2 % Hh; pos = h*Wd + w; }   // (w,h) scan
  else pos = l2;                                                 // (h,w) scan
  xs_h[idx] = (_Float16)xcs[((size_t)(b*HW+pos))*Dd + d];        // coalesced
}

// ---- K7: x_proj batched GEMM per k: [9216,192]x[48,192]^T -----------------
__global__ void k_xproj(const _Float16* __restrict__ xs_h, const _Float16* __restrict__ W,
                        float* xdbl){
  int lane = threadIdx.x & 31;
  int k   = blockIdx.z;
  int tM4 = blockIdx.x*4 + (threadIdx.x>>5);    // 64-row group
  int tN  = blockIdx.y;
  const _Float16* A  = xs_h + (size_t)k*BHW*Dd;
  const _Float16* Wk = W    + (size_t)k*NPAD*Dd;
  ACC4_INIT;
  wmma_rowcol4(A + (size_t)tM4*64*Dd, Wk + (size_t)tN*16*Dd, Dd, Dd, Dd, lane, acc);
  int n = tN*16 + (lane&15);
  if (n >= NCOLS) return;                       // padded cols 38..47 dropped
  int roff = (lane<16)?0:8;
  #pragma unroll
  for (int t=0;t<4;t++){
    int mbase = tM4*64 + t*16 + roff;
    #pragma unroll
    for (int r=0;r<8;r++)
      xdbl[((size_t)k*BHW + mbase + r)*NCOLS + n] = acc[t][r];
  }
}

// ===========================================================================
// Chunked selective scan: h' = exp(dt*A)h + dt*u*B  (linear recurrence)
// S1: per-chunk local scan from h=0 -> h_end + sum(dt); S2: sequential chunk
// combine (transfer = exp(A*sum_dt), exact); S3: replay with correct h_in.
// 16 blocks -> 256 blocks of parallelism on the serial critical path.
// ===========================================================================
__global__ void k_scan_p1(const float* __restrict__ xdbl, const _Float16* __restrict__ xs_h,
                          const float* __restrict__ dt_w, const float* __restrict__ dt_b,
                          const float* __restrict__ A_logs,
                          float* hend, float* sdt){
  int blk = blockIdx.x;                         // (b*Kk + k)*NCH + c
  int c  = blk % NCH;  int bk = blk / NCH;
  int b  = bk / Kk;    int k  = bk % Kk;
  int d  = threadIdx.x;
  int kd = k*Dd + d;
  float dtw[DTR];
  #pragma unroll
  for (int r=0;r<DTR;r++) dtw[r] = dt_w[kd*DTR + r];
  float dtb = dt_b[kd];
  float Ar[NST];
  #pragma unroll
  for (int n=0;n<NST;n++) Ar[n] = -__expf(A_logs[kd*NST + n]);
  float h[NST];
  #pragma unroll
  for (int n=0;n<NST;n++) h[n] = 0.f;
  float S = 0.f;
  const float*    xb = xdbl + (size_t)(k*BHW + b*Ll + c*CLEN)*NCOLS;
  const _Float16* ub = xs_h + ((size_t)(k*Bn+b)*Ll + c*CLEN)*Dd + d;
  for (int l=0; l<CLEN; l++){
    if (l+8 < CLEN){
      __builtin_prefetch(xb + (size_t)(l+8)*NCOLS, 0, 1);  // global_prefetch_b8
      __builtin_prefetch(ub + (size_t)(l+8)*Dd,    0, 1);
    }
    const float* xr = xb + (size_t)l*NCOLS;
    float u = (float)ub[(size_t)l*Dd];
    float dtr = dtb;
    #pragma unroll
    for (int r=0;r<DTR;r++) dtr += dtw[r]*xr[r];
    float dtv = softplusf_(dtr);
    S += dtv;
    float du = dtv*u;
    #pragma unroll
    for (int n=0;n<NST;n++)
      h[n] = h[n]*__expf(dtv*Ar[n]) + du*xr[6+n];
  }
  size_t base = ((size_t)blk*Dd + d)*NST;
  #pragma unroll
  for (int n=0;n<NST;n++) hend[base+n] = h[n];
  sdt[(size_t)blk*Dd + d] = S;
}

__global__ void k_scan_p2(const float* __restrict__ A_logs,
                          float* hend, const float* __restrict__ sdt){
  int bk = blockIdx.x;                          // b*Kk + k
  int k  = bk % Kk;
  int d  = threadIdx.x;
  int kd = k*Dd + d;
  float Ar[NST];
  #pragma unroll
  for (int n=0;n<NST;n++) Ar[n] = -__expf(A_logs[kd*NST + n]);
  float hin[NST];
  #pragma unroll
  for (int n=0;n<NST;n++) hin[n] = 0.f;
  for (int c=0; c<NCH; c++){
    size_t blk = (size_t)bk*NCH + c;
    float S = sdt[blk*Dd + d];
    size_t base = (blk*Dd + d)*NST;
    #pragma unroll
    for (int n=0;n<NST;n++){
      float he = hend[base+n];
      float nh = __expf(Ar[n]*S)*hin[n] + he;   // exact chunk transfer
      hend[base+n] = hin[n];                    // rewrite with chunk h_in
      hin[n] = nh;
    }
  }
}

__global__ void k_scan_p3(const float* __restrict__ xdbl, const _Float16* __restrict__ xs_h,
                          const float* __restrict__ dt_w, const float* __restrict__ dt_b,
                          const float* __restrict__ A_logs, const float* __restrict__ Ds,
                          const float* __restrict__ hend, float* ys){
  int blk = blockIdx.x;
  int c  = blk % NCH;  int bk = blk / NCH;
  int b  = bk / Kk;    int k  = bk % Kk;
  int d  = threadIdx.x;
  int kd = k*Dd + d;
  float dtw[DTR];
  #pragma unroll
  for (int r=0;r<DTR;r++) dtw[r] = dt_w[kd*DTR + r];
  float dtb = dt_b[kd];
  float Ar[NST];
  #pragma unroll
  for (int n=0;n<NST;n++) Ar[n] = -__expf(A_logs[kd*NST + n]);
  float dsv = Ds[kd];
  size_t base = ((size_t)blk*Dd + d)*NST;
  float h[NST];
  #pragma unroll
  for (int n=0;n<NST;n++) h[n] = hend[base+n];  // incoming state
  const float*    xb = xdbl + (size_t)(k*BHW + b*Ll + c*CLEN)*NCOLS;
  const _Float16* ub = xs_h + ((size_t)(k*Bn+b)*Ll + c*CLEN)*Dd + d;
  float*          yb = ys   + ((size_t)(b*Kk+k)*Ll + c*CLEN)*Dd + d;
  for (int l=0; l<CLEN; l++){
    if (l+8 < CLEN){
      __builtin_prefetch(xb + (size_t)(l+8)*NCOLS, 0, 1);
      __builtin_prefetch(ub + (size_t)(l+8)*Dd,    0, 1);
    }
    const float* xr = xb + (size_t)l*NCOLS;
    float u = (float)ub[(size_t)l*Dd];
    float dtr = dtb;
    #pragma unroll
    for (int r=0;r<DTR;r++) dtr += dtw[r]*xr[r];
    float dtv = softplusf_(dtr);
    float du = dtv*u;
    float y = 0.f;
    #pragma unroll
    for (int n=0;n<NST;n++){
      h[n] = h[n]*__expf(dtv*Ar[n]) + du*xr[6+n];
      y   += h[n]*xr[22+n];
    }
    yb[(size_t)l*Dd] = y + dsv*u;               // coalesced over d
  }
}

// ---- K9: combine 4 directions + LN(D) + gate by silu(z) -> f16 ------------
__global__ void k_combine(const float* __restrict__ ys, const float* __restrict__ zs,
                          const float* __restrict__ og, const float* __restrict__ ob,
                          _Float16* yg_h){
  __shared__ float lsum[6], lsq[6], stats[2];
  int row = blockIdx.x;                 // b*HW + hw
  int d   = threadIdx.x;
  int b = row / HW, hw = row % HW;
  int h = hw / Wd, w = hw % Wd;
  int lwh = w*Hh + h;
  size_t base = (size_t)b*Kk*Ll*Dd;
  float y = ys[base + ((size_t)0*Ll + hw)        *Dd + d]
          + ys[base + ((size_t)2*Ll + (Ll-1-hw)) *Dd + d]
          + ys[base + ((size_t)1*Ll + lwh)       *Dd + d]
          + ys[base + ((size_t)3*Ll + (Ll-1-lwh))*Dd + d];
  float s1 = y, s2 = y*y;
  for (int o=16;o>0;o>>=1){ s1+=__shfl_xor(s1,o,32); s2+=__shfl_xor(s2,o,32); }
  int wid = threadIdx.x>>5;
  if ((threadIdx.x&31)==0){ lsum[wid]=s1; lsq[wid]=s2; }
  __syncthreads();
  if (threadIdx.x==0){
    float a1=0.f,a2=0.f;
    for (int i=0;i<6;i++){ a1+=lsum[i]; a2+=lsq[i]; }
    float m = a1*(1.f/Dd);
    stats[0]=m; stats[1]=rsqrtf(a2*(1.f/Dd)-m*m+1e-5f);
  }
  __syncthreads();
  float yn = (y - stats[0])*stats[1]*og[d] + ob[d];
  yg_h[(size_t)row*Dd + d] = (_Float16)(yn * zs[(size_t)row*Dd + d]);
}

// ---- K10: out_proj GEMM [9216,192]x[96,192]^T + residual + NCHW store -----
__global__ void k_outproj(const _Float16* __restrict__ A, const _Float16* __restrict__ W,
                          const float* __restrict__ x, const float* __restrict__ att,
                          float* out){
  int lane = threadIdx.x & 31;
  int tM4 = blockIdx.x*4 + (threadIdx.x>>5);    // 64-row group
  int tN  = blockIdx.y;
  ACC4_INIT;
  wmma_rowcol4(A + (size_t)tM4*64*Dd, W + (size_t)tN*16*Dd, Dd, Dd, Dd, lane, acc);
  int n = tN*16 + (lane&15);
  int roff = (lane<16)?0:8;
  float av = att[0];  // placeholder to keep loads simple per m below
  (void)av;
  #pragma unroll
  for (int t=0;t<4;t++){
    int mbase = tM4*64 + t*16 + roff;
    #pragma unroll
    for (int r=0;r<8;r++){
      int m = mbase + r;
      int b = m/HW, hw = m%HW;
      size_t oi = ((size_t)(b*Cc+n))*HW + hw;
      out[oi] = acc[t][r] + x[oi]*att[b*Cc+n];  // residual xi = x*att, fused
    }
  }
}

// ---------------------------------------------------------------------------
extern "C" void kernel_launch(void* const* d_in, const int* in_sizes, int n_in,
                              void* d_out, int out_size, void* d_ws, size_t ws_size,
                              hipStream_t stream){
  const float* x      = (const float*)d_in[0];
  const float* ca_w1  = (const float*)d_in[1];
  const float* ca_w2  = (const float*)d_in[2];
  const float* ln_g   = (const float*)d_in[3];
  const float* ln_b   = (const float*)d_in[4];
  const float* in_w   = (const float*)d_in[5];
  const float* conv_w = (const float*)d_in[6];
  const float* conv_b = (const float*)d_in[7];
  const float* xp_w   = (const float*)d_in[8];
  const float* dt_w   = (const float*)d_in[9];
  const float* dt_b   = (const float*)d_in[10];
  const float* A_logs = (const float*)d_in[11];
  const float* Ds     = (const float*)d_in[12];
  const float* on_g   = (const float*)d_in[13];
  const float* on_b   = (const float*)d_in[14];
  const float* out_w  = (const float*)d_in[15];
  float* out = (float*)d_out;

  char* ws = (char*)d_ws;
  size_t off = 0;
  auto alloc = [&](size_t bytes)->char*{
    char* p = ws + off; off = (off + bytes + 255) & ~(size_t)255; return p;
  };
  float*    s      = (float*)   alloc((size_t)Bn*Cc*4);
  float*    att    = (float*)   alloc((size_t)Bn*Cc*4);
  _Float16* xn_h   = (_Float16*)alloc((size_t)BHW*Cc*2);
  _Float16* inw_h  = (_Float16*)alloc((size_t)2*Dd*Cc*2);
  _Float16* xpw_h  = (_Float16*)alloc((size_t)Kk*NPAD*Dd*2);
  _Float16* opw_h  = (_Float16*)alloc((size_t)Cc*Dd*2);
  float*    xc_pre = (float*)   alloc((size_t)Bn*HW*Dd*4);
  float*    zs     = (float*)   alloc((size_t)BHW*Dd*4);
  float*    xcs    = (float*)   alloc((size_t)Bn*HW*Dd*4);
  _Float16* xs_h   = (_Float16*)alloc((size_t)Kk*BHW*Dd*2);
  float*    xdbl   = (float*)   alloc((size_t)Kk*BHW*NCOLS*4);
  float*    ys     = (float*)   alloc((size_t)Bn*Kk*Ll*Dd*4);
  _Float16* yg_h   = (_Float16*)alloc((size_t)BHW*Dd*2);
  float*    hend   = (float*)   alloc((size_t)Bn*Kk*NCH*Dd*NST*4);
  float*    sdt    = (float*)   alloc((size_t)Bn*Kk*NCH*Dd*4);

  k_wpack  <<<144, 256, 0, stream>>>(in_w, xp_w, out_w, inw_h, xpw_h, opw_h);
  k_mean   <<<Bn*Cc, 256, 0, stream>>>(x, s);
  k_camlp  <<<1, 128, 0, stream>>>(s, ca_w1, ca_w2, att);
  k_ln_pack<<<BHW/4, 128, 0, stream>>>(x, att, ln_g, ln_b, xn_h);
  k_inproj <<<dim3(BHW/64, (2*Dd)/64), 128, 0, stream>>>(xn_h, inw_h, xc_pre, zs);
  k_dwconv <<<(Bn*HW*Dd + 255)/256, 256, 0, stream>>>(xc_pre, conv_w, conv_b, xcs);
  k_pack_xs<<<(int)(((size_t)Kk*Bn*Ll*Dd + 255)/256), 256, 0, stream>>>(xcs, xs_h);
  k_xproj  <<<dim3(BHW/256, NPAD/16, Kk), 128, 0, stream>>>(xs_h, xpw_h, xdbl);
  k_scan_p1<<<Bn*Kk*NCH, Dd, 0, stream>>>(xdbl, xs_h, dt_w, dt_b, A_logs, hend, sdt);
  k_scan_p2<<<Bn*Kk, Dd, 0, stream>>>(A_logs, hend, sdt);
  k_scan_p3<<<Bn*Kk*NCH, Dd, 0, stream>>>(xdbl, xs_h, dt_w, dt_b, A_logs, Ds, hend, ys);
  k_combine<<<BHW, Dd, 0, stream>>>(ys, zs, on_g, on_b, yg_h);
  k_outproj<<<dim3(BHW/256, Cc/16), 128, 0, stream>>>(yg_h, opw_h, x, att, out);
}
